// MixtralMoE_88785563943402
// MI455X (gfx1250) — matrix operand order repaired and checked by reading, not compiled, not run on hardware
//
#include <hip/hip_runtime.h>

#define NUM_EXPERTS 8
#define HIDDEN 2048
#define INTER 4096
#define TOKENS 8192

typedef __attribute__((ext_vector_type(16))) __bf16        v16bf;
typedef __attribute__((ext_vector_type(8)))  float         v8f;
typedef __attribute__((ext_vector_type(4))) unsigned int   u32x4;
typedef __attribute__((ext_vector_type(2))) unsigned int   u32x2;

union Frag { u32x4 q[2]; v16bf v; };

__device__ __forceinline__ unsigned int f2bf_bits(float f) {
    union { float f; unsigned int u; } x; x.f = f;
    return (x.u + 0x7FFFu + ((x.u >> 16) & 1u)) >> 16;  // RNE truncate
}

__device__ __forceinline__ v8f zero_v8f() {
    v8f r;
    for (int i = 0; i < 8; ++i) r[i] = 0.0f;
    return r;
}

// ---------------------------------------------------------------------------
// CDNA5 async global -> LDS (ASYNCcnt-tracked; no VGPR transit).
// LDS dest offset = low 32 bits of the flat address of a __shared__ object
// (aperture mapping: LDS offset is address bits 31:0, wave-relative).
// ---------------------------------------------------------------------------
__device__ __forceinline__ unsigned lds_off(const void* p) {
    return (unsigned)(unsigned long long)(uintptr_t)p;
}
__device__ __forceinline__ void async_load_b128(unsigned ldsaddr, unsigned long long gaddr) {
    asm volatile("global_load_async_to_lds_b128 %0, %1, off"
                 :: "v"(ldsaddr), "v"(gaddr) : "memory");
}
#if __has_builtin(__builtin_amdgcn_s_wait_asynccnt)
#define WAIT_ASYNCCNT(n) __builtin_amdgcn_s_wait_asynccnt(n)
#else
#define WAIT_ASYNCCNT(n) asm volatile("s_wait_asynccnt %0" :: "i"(n) : "memory")
#endif

// ---------------------------------------------------------------------------
// Zero output + per-expert counters (runs first every call)
// ---------------------------------------------------------------------------
__global__ __launch_bounds__(256)
void zero_kernel(float* __restrict__ out, int n, int* __restrict__ counts) {
    int i = blockIdx.x * blockDim.x + threadIdx.x;
    if (i < NUM_EXPERTS) counts[i] = 0;
    int stride = gridDim.x * blockDim.x;
    for (; i < n; i += stride) out[i] = 0.0f;
}

// ---------------------------------------------------------------------------
// Router: logits -> softmax -> top-2 -> renorm; writes combine [T,8] AND
// appends token to its two experts' compacted lists.
// ---------------------------------------------------------------------------
__global__ __launch_bounds__(256)
void router_kernel(const float* __restrict__ x,
                   const float* __restrict__ wg,
                   float* __restrict__ combine,
                   int* __restrict__ counts,
                   int* __restrict__ tokenList) {
    __shared__ float wgs[NUM_EXPERTS * HIDDEN]; // 64 KB
    for (int i = threadIdx.x; i < NUM_EXPERTS * HIDDEN; i += blockDim.x)
        wgs[i] = wg[i];
    __syncthreads();

    const int t = blockIdx.x * blockDim.x + threadIdx.x;
    float acc[NUM_EXPERTS];
    #pragma unroll
    for (int e = 0; e < NUM_EXPERTS; ++e) acc[e] = 0.0f;

    const float4* xr = (const float4*)(x + (size_t)t * HIDDEN);
    for (int k4 = 0; k4 < HIDDEN / 4; ++k4) {
        float4 xv = xr[k4];
        #pragma unroll
        for (int e = 0; e < NUM_EXPERTS; ++e) {
            float4 wv = ((const float4*)(wgs + e * HIDDEN))[k4];
            acc[e] += xv.x * wv.x + xv.y * wv.y + xv.z * wv.z + xv.w * wv.w;
        }
    }
    float m = acc[0];
    #pragma unroll
    for (int e = 1; e < NUM_EXPERTS; ++e) m = fmaxf(m, acc[e]);
    float p[NUM_EXPERTS], s = 0.0f;
    #pragma unroll
    for (int e = 0; e < NUM_EXPERTS; ++e) { p[e] = __expf(acc[e] - m); s += p[e]; }
    float inv = 1.0f / s;
    #pragma unroll
    for (int e = 0; e < NUM_EXPERTS; ++e) p[e] *= inv;

    int i1 = 0; float v1 = p[0];
    #pragma unroll
    for (int e = 1; e < NUM_EXPERTS; ++e) if (p[e] > v1) { v1 = p[e]; i1 = e; }
    int i2 = -1; float v2 = -1.0f;
    #pragma unroll
    for (int e = 0; e < NUM_EXPERTS; ++e) if (e != i1 && p[e] > v2) { v2 = p[e]; i2 = e; }
    float rn = 1.0f / (v1 + v2);
    #pragma unroll
    for (int e = 0; e < NUM_EXPERTS; ++e) {
        float w = (e == i1) ? v1 * rn : (e == i2 ? v2 * rn : 0.0f);
        combine[(size_t)t * NUM_EXPERTS + e] = w;
    }
    int s1 = atomicAdd(counts + i1, 1);
    tokenList[i1 * TOKENS + s1] = t;
    int s2 = atomicAdd(counts + i2, 1);
    tokenList[i2 * TOKENS + s2] = t;
}

// ---------------------------------------------------------------------------
// fp32 -> bf16 bulk convert, 4 elems/thread/iter (n % 4 == 0)
// ---------------------------------------------------------------------------
__global__ __launch_bounds__(256)
void cvt_bf16_kernel(const float* __restrict__ in, unsigned short* __restrict__ out, int n) {
    int i = blockIdx.x * blockDim.x + threadIdx.x;
    int stride = gridDim.x * blockDim.x;
    const int n4 = n >> 2;
    for (; i < n4; i += stride) {
        float4 v = ((const float4*)in)[i];
        u32x2 pk;
        pk[0] = f2bf_bits(v.x) | (f2bf_bits(v.y) << 16);
        pk[1] = f2bf_bits(v.z) | (f2bf_bits(v.w) << 16);
        *(u32x2*)(out + (size_t)i * 4) = pk;
    }
}

// ---------------------------------------------------------------------------
// Gathered GEMM1 + SwiGLU, async-LDS double-buffered.
//   Hb[slot,:] = silu(x[tok] @ W1g^T) * (x[tok] @ W1u^T),  slot < count[e]
// block 256 thr = 8 waves; wave tile 64x32; block tile 128x128; k-step 32.
// Per k-step each thread issues 6 async b128 (A,Bg,Bu tiles = 24KB/block).
// ---------------------------------------------------------------------------
__global__ __launch_bounds__(256)
void gemm1_silu_kernel(const unsigned short* __restrict__ Xb,   // [TOKENS, HIDDEN]
                       const unsigned short* __restrict__ W1b,  // [2*INTER, HIDDEN]
                       unsigned short* __restrict__ Hb,         // [TOKENS, INTER] (slots)
                       const int* __restrict__ tokenList,       // [TOKENS] this expert
                       const int* __restrict__ counts,
                       int expert) {
    __shared__ unsigned short ldsA[2][128 * 32];   // 16 KB
    __shared__ unsigned short ldsBg[2][128 * 32];  // 16 KB
    __shared__ unsigned short ldsBu[2][128 * 32];  // 16 KB

    const int cnt = counts[expert];
    if ((int)blockIdx.x * 128 >= cnt) return;   // uniform block early-exit

    const int tid   = threadIdx.x;
    const int lane  = tid & 31;
    const int wave  = tid >> 5;
    const int waveM = wave >> 2;   // 0..1
    const int waveN = wave & 3;    // 0..3
    const int lm = lane & 15;
    const int lh = lane >> 4;

    // --- staging assignment: 2 threads per tile row, 2 b128 chunks each ---
    const int sr = tid >> 1;               // tile row 0..127
    const int sc = (tid & 1) * 32;         // byte offset within 64B row-chunk
    int slot = blockIdx.x * 128 + sr;
    slot = (slot < cnt) ? slot : (cnt - 1);
    const unsigned long long aSrc =
        (unsigned long long)(uintptr_t)(Xb + (size_t)tokenList[slot] * HIDDEN) + sc;
    const unsigned long long gSrc =
        (unsigned long long)(uintptr_t)(W1b + (size_t)(blockIdx.y * 128 + sr) * HIDDEN) + sc;
    const unsigned long long uSrc =
        (unsigned long long)(uintptr_t)(W1b + (size_t)(INTER + blockIdx.y * 128 + sr) * HIDDEN) + sc;
    const unsigned dstRow = (unsigned)(sr * 64 + sc);
    const unsigned ldsAo[2]  = { lds_off(&ldsA[0][0]),  lds_off(&ldsA[1][0])  };
    const unsigned ldsBgo[2] = { lds_off(&ldsBg[0][0]), lds_off(&ldsBg[1][0]) };
    const unsigned ldsBuo[2] = { lds_off(&ldsBu[0][0]), lds_off(&ldsBu[1][0]) };

    auto issue_stage = [&](int buf, int k) {
        const unsigned long long kB = (unsigned long long)k * 2;
        async_load_b128(ldsAo[buf]  + dstRow,      aSrc + kB);
        async_load_b128(ldsAo[buf]  + dstRow + 16, aSrc + kB + 16);
        async_load_b128(ldsBgo[buf] + dstRow,      gSrc + kB);
        async_load_b128(ldsBgo[buf] + dstRow + 16, gSrc + kB + 16);
        async_load_b128(ldsBuo[buf] + dstRow,      uSrc + kB);
        async_load_b128(ldsBuo[buf] + dstRow + 16, uSrc + kB + 16);
    };

    v8f accg[4][2], accu[4][2];
    #pragma unroll
    for (int i = 0; i < 4; ++i)
        #pragma unroll
        for (int j = 0; j < 2; ++j) { accg[i][j] = zero_v8f(); accu[i][j] = zero_v8f(); }

    const int NK = HIDDEN / 32;   // 64
    issue_stage(0, 0);
    for (int i = 0; i < NK; ++i) {
        const int cur = i & 1;
        if (i + 1 < NK) {
            issue_stage(1 - cur, (i + 1) * 32);
            WAIT_ASYNCCNT(6);        // oldest 6 (stage cur) complete
        } else {
            WAIT_ASYNCCNT(0);
        }
        __syncthreads();             // stage cur visible to all waves

        Frag A[4];
        #pragma unroll
        for (int mf = 0; mf < 4; ++mf) {
            const int row = waveM * 64 + mf * 16 + lm;
            A[mf].q[0] = *(const u32x4*)&ldsA[cur][row * 32 + lh * 16];
            A[mf].q[1] = *(const u32x4*)&ldsA[cur][row * 32 + lh * 16 + 8];
        }
        Frag Bg[2], Bu[2];
        #pragma unroll
        for (int nf = 0; nf < 2; ++nf) {
            const int row = waveN * 32 + nf * 16 + lm;
            Bg[nf].q[0] = *(const u32x4*)&ldsBg[cur][row * 32 + lh * 16];
            Bg[nf].q[1] = *(const u32x4*)&ldsBg[cur][row * 32 + lh * 16 + 8];
            Bu[nf].q[0] = *(const u32x4*)&ldsBu[cur][row * 32 + lh * 16];
            Bu[nf].q[1] = *(const u32x4*)&ldsBu[cur][row * 32 + lh * 16 + 8];
        }
        #pragma unroll
        for (int mf = 0; mf < 4; ++mf)
            #pragma unroll
            for (int nf = 0; nf < 2; ++nf) {
                accg[mf][nf] = __builtin_amdgcn_wmma_f32_16x16x32_bf16(
                    false, A[mf].v, false, Bg[nf].v, (short)0, accg[mf][nf], false, false);
                accu[mf][nf] = __builtin_amdgcn_wmma_f32_16x16x32_bf16(
                    false, A[mf].v, false, Bu[nf].v, (short)0, accu[mf][nf], false, false);
            }
        __syncthreads();             // reads done before buf is overwritten
    }

    const int mBase = blockIdx.x * 128 + waveM * 64;
    const int nBase = blockIdx.y * 128 + waveN * 32;
    #pragma unroll
    for (int mf = 0; mf < 4; ++mf) {
        #pragma unroll
        for (int nf = 0; nf < 2; ++nf) {
            const int col = nBase + nf * 16 + lm;
            #pragma unroll
            for (int r = 0; r < 8; ++r) {
                const int s = mBase + mf * 16 + r + lh * 8;
                if (s < cnt) {
                    float g = accg[mf][nf][r];
                    float u = accu[mf][nf][r];
                    float h = (g / (1.0f + __expf(-g))) * u;
                    Hb[(size_t)s * INTER + col] = (unsigned short)f2bf_bits(h);
                }
            }
        }
    }
}

// ---------------------------------------------------------------------------
// Gathered GEMM2 + scaled scatter-accumulate, async-LDS double-buffered.
//   out[tok] += combine[tok, e] * (Hb[slot] @ W2^T)
// ---------------------------------------------------------------------------
__global__ __launch_bounds__(256)
void gemm2_kernel(const unsigned short* __restrict__ Hb,   // [TOKENS, INTER] (slots)
                  const unsigned short* __restrict__ W2b,  // [HIDDEN, INTER]
                  const float* __restrict__ combine,       // [TOKENS, 8]
                  const int* __restrict__ tokenList,       // [TOKENS] this expert
                  const int* __restrict__ counts,
                  float* __restrict__ out,                 // [TOKENS, HIDDEN]
                  int expert) {
    __shared__ unsigned short ldsA[2][128 * 32];  // 16 KB
    __shared__ unsigned short ldsB[2][128 * 32];  // 16 KB

    const int cnt = counts[expert];
    if ((int)blockIdx.x * 128 >= cnt) return;

    const int tid   = threadIdx.x;
    const int lane  = tid & 31;
    const int wave  = tid >> 5;
    const int waveM = wave >> 2;
    const int waveN = wave & 3;
    const int lm = lane & 15;
    const int lh = lane >> 4;

    const int sr = tid >> 1;
    const int sc = (tid & 1) * 32;
    const unsigned long long aSrc =
        (unsigned long long)(uintptr_t)(Hb + (size_t)(blockIdx.x * 128 + sr) * INTER) + sc;
    const unsigned long long bSrc =
        (unsigned long long)(uintptr_t)(W2b + (size_t)(blockIdx.y * 128 + sr) * INTER) + sc;
    const unsigned dstRow = (unsigned)(sr * 64 + sc);
    const unsigned ldsAo[2] = { lds_off(&ldsA[0][0]), lds_off(&ldsA[1][0]) };
    const unsigned ldsBo[2] = { lds_off(&ldsB[0][0]), lds_off(&ldsB[1][0]) };

    auto issue_stage = [&](int buf, int k) {
        const unsigned long long kB = (unsigned long long)k * 2;
        async_load_b128(ldsAo[buf] + dstRow,      aSrc + kB);
        async_load_b128(ldsAo[buf] + dstRow + 16, aSrc + kB + 16);
        async_load_b128(ldsBo[buf] + dstRow,      bSrc + kB);
        async_load_b128(ldsBo[buf] + dstRow + 16, bSrc + kB + 16);
    };

    v8f acc[4][2];
    #pragma unroll
    for (int i = 0; i < 4; ++i)
        #pragma unroll
        for (int j = 0; j < 2; ++j) acc[i][j] = zero_v8f();

    const int NK = INTER / 32;   // 128
    issue_stage(0, 0);
    for (int i = 0; i < NK; ++i) {
        const int cur = i & 1;
        if (i + 1 < NK) {
            issue_stage(1 - cur, (i + 1) * 32);
            WAIT_ASYNCCNT(4);
        } else {
            WAIT_ASYNCCNT(0);
        }
        __syncthreads();

        Frag A[4];
        #pragma unroll
        for (int mf = 0; mf < 4; ++mf) {
            const int row = waveM * 64 + mf * 16 + lm;
            A[mf].q[0] = *(const u32x4*)&ldsA[cur][row * 32 + lh * 16];
            A[mf].q[1] = *(const u32x4*)&ldsA[cur][row * 32 + lh * 16 + 8];
        }
        Frag B[2];
        #pragma unroll
        for (int nf = 0; nf < 2; ++nf) {
            const int row = waveN * 32 + nf * 16 + lm;
            B[nf].q[0] = *(const u32x4*)&ldsB[cur][row * 32 + lh * 16];
            B[nf].q[1] = *(const u32x4*)&ldsB[cur][row * 32 + lh * 16 + 8];
        }
        #pragma unroll
        for (int mf = 0; mf < 4; ++mf)
            #pragma unroll
            for (int nf = 0; nf < 2; ++nf)
                acc[mf][nf] = __builtin_amdgcn_wmma_f32_16x16x32_bf16(
                    false, A[mf].v, false, B[nf].v, (short)0, acc[mf][nf], false, false);
        __syncthreads();
    }

    const int mBase = blockIdx.x * 128 + waveM * 64;
    const int nBase = blockIdx.y * 128 + waveN * 32;
    #pragma unroll
    for (int mf = 0; mf < 4; ++mf) {
        #pragma unroll
        for (int r = 0; r < 8; ++r) {
            const int s = mBase + mf * 16 + r + lh * 8;
            if (s < cnt) {
                const int tok = tokenList[s];
                const float scale = combine[(size_t)tok * NUM_EXPERTS + expert];
                #pragma unroll
                for (int nf = 0; nf < 2; ++nf) {
                    const int col = nBase + nf * 16 + lm;
                    const size_t o = (size_t)tok * HIDDEN + col;
                    out[o] += scale * acc[mf][nf][r];  // one add per (tok,e); serialized
                }
            }
        }
    }
}

// ---------------------------------------------------------------------------
// Launch: zero -> router -> cvt X -> per expert { cvt W1, gemm1+silu, cvt W2, gemm2 }
// ---------------------------------------------------------------------------
extern "C" void kernel_launch(void* const* d_in, const int* in_sizes, int n_in,
                              void* d_out, int out_size, void* d_ws, size_t ws_size,
                              hipStream_t stream) {
    const float* x  = (const float*)d_in[0];
    const float* wg = (const float*)d_in[1];
    const float* w1 = (const float*)d_in[2];
    const float* w2 = (const float*)d_in[3];
    float* out = (float*)d_out;
    (void)in_sizes; (void)n_in; (void)out_size; (void)ws_size;

    char* ws = (char*)d_ws;
    size_t off = 0;
    float*          combine   = (float*)(ws + off);          off += (size_t)TOKENS * NUM_EXPERTS * 4;
    int*            counts    = (int*)(ws + off);            off += 1024;
    int*            tokenList = (int*)(ws + off);            off += (size_t)NUM_EXPERTS * TOKENS * 4;
    unsigned short* Xb        = (unsigned short*)(ws + off); off += (size_t)TOKENS * HIDDEN * 2;
    unsigned short* Hb        = (unsigned short*)(ws + off); off += (size_t)TOKENS * INTER * 2;
    unsigned short* W1b       = (unsigned short*)(ws + off); off += (size_t)2 * INTER * HIDDEN * 2;
    unsigned short* W2b       = (unsigned short*)(ws + off);

    zero_kernel<<<2048, 256, 0, stream>>>(out, TOKENS * HIDDEN, counts);
    router_kernel<<<TOKENS / 256, 256, 0, stream>>>(x, wg, combine, counts, tokenList);
    cvt_bf16_kernel<<<2048, 256, 0, stream>>>(x, Xb, TOKENS * HIDDEN);

    dim3 g1(TOKENS / 128, INTER / 128);   // 64 x 32 (M blocks early-exit past count)
    dim3 g2(TOKENS / 128, HIDDEN / 128);  // 64 x 16
    for (int e = 0; e < NUM_EXPERTS; ++e) {
        cvt_bf16_kernel<<<2048, 256, 0, stream>>>(w1 + (size_t)e * 2 * INTER * HIDDEN, W1b,
                                                  2 * INTER * HIDDEN);
        gemm1_silu_kernel<<<g1, 256, 0, stream>>>(Xb, W1b, Hb, tokenList + e * TOKENS,
                                                  counts, e);
        cvt_bf16_kernel<<<2048, 256, 0, stream>>>(w2 + (size_t)e * HIDDEN * INTER, W2b,
                                                  HIDDEN * INTER);
        gemm2_kernel<<<g2, 256, 0, stream>>>(Hb, W2b, combine, tokenList + e * TOKENS,
                                             counts, out, e);
    }
}
